// ClimbEGNNDiffusionTrainer_32246614459258
// MI455X (gfx1250) — matrix operand order repaired and checked by reading, not compiled, not run on hardware
//
#include <hip/hip_runtime.h>

#define GG 512
#define PP 30
#define KK 12
#define HH 128
#define LL 6
#define NN (GG*PP)
#define NSS 6
#define NRR 5
#define NULLR 4

typedef __attribute__((ext_vector_type(16))) _Float16 v16h;
typedef __attribute__((ext_vector_type(8)))  _Float16 v8h;
typedef __attribute__((ext_vector_type(8)))  float    v8f;

__device__ __forceinline__ float silu_f(float x){ return x / (1.0f + __expf(-x)); }

// ---- WMMA fragment loads (CDNA5 wave32 layouts, CDNA5 ISA) ----
// A: 16x32 f16 tile, row-major in LDS. lanes 0-15: K0..7 + K16..23 ; lanes 16-31: K8..15 + K24..31
__device__ __forceinline__ v16h frag_a(const _Float16* tile, int stride, int k0, int lane){
  const int row = lane & 15;
  const int hs  = (lane >> 4) & 1;
  const _Float16* p = tile + row*stride + k0 + hs*8;
  v8h lo = *(const v8h*)(p);
  v8h hi = *(const v8h*)(p + 16);
  v16h a;
#pragma unroll
  for (int i=0;i<8;++i){ a[i]=lo[i]; a[8+i]=hi[i]; }
  return a;
}
// B: 32x16 f16 tile from pre-transposed weights WT[out][Kpad] (contiguous in K).
// lanes 0-15 hold K=k0..k0+7 + k0+16..k0+23 for col n0+lane ; lanes 16-31 hold K=k0+8..k0+15 + k0+24..k0+31
__device__ __forceinline__ v16h frag_b(const _Float16* wt, int Kpad, int n0, int k0, int lane){
  const int n  = n0 + (lane & 15);
  const int ks = (lane >> 4) & 1;
  const _Float16* p = wt + n*Kpad + k0 + ks*8;
  v8h lo = *(const v8h*)(p);
  v8h hi = *(const v8h*)(p + 16);
  v16h b;
#pragma unroll
  for (int i=0;i<8;++i){ b[i]=lo[i]; b[8+i]=hi[i]; }
  return b;
}

// 16xKin @ WT(Kin x 128) + bias [-> silu] -> f16 LDS tile (16x128, strideO)
__device__ __forceinline__ void gemm_lds(const _Float16* A, int strideA, int ksteps,
                                         const _Float16* WT, int Kpad,
                                         const float* bias, bool do_silu,
                                         _Float16* out, int strideO, int lane)
{
  for (int nt=0; nt<8; ++nt){
    v8f c = {0.f,0.f,0.f,0.f,0.f,0.f,0.f,0.f};
    for (int kk=0; kk<ksteps; ++kk){
      v16h a = frag_a(A, strideA, kk*32, lane);
      v16h b = frag_b(WT, Kpad, nt*16, kk*32, lane);
      c = __builtin_amdgcn_wmma_f32_16x16x32_f16(false, a, false, b, (short)0, c, false, false);
    }
    const int n = nt*16 + (lane & 15);
    const float bv = bias[n];
    const int mbase = (lane < 16) ? 0 : 8;
#pragma unroll
    for (int r=0;r<8;++r){
      float y = c[r] + bv;
      if (do_silu) y = silu_f(y);
      out[(mbase + r)*strideO + n] = (_Float16)y;
    }
  }
}

// 16xKin @ WT + bias [+resid] -> f32 global rows of width HH
__device__ __forceinline__ void gemm_glb(const _Float16* A, int strideA, int ksteps,
                                         const _Float16* WT, int Kpad,
                                         const float* bias,
                                         const float* resid, float* out, int lane)
{
  for (int nt=0; nt<8; ++nt){
    v8f c = {0.f,0.f,0.f,0.f,0.f,0.f,0.f,0.f};
    for (int kk=0; kk<ksteps; ++kk){
      v16h a = frag_a(A, strideA, kk*32, lane);
      v16h b = frag_b(WT, Kpad, nt*16, kk*32, lane);
      c = __builtin_amdgcn_wmma_f32_16x16x32_f16(false, a, false, b, (short)0, c, false, false);
    }
    const int n = nt*16 + (lane & 15);
    const float bv = bias[n];
    const int mbase = (lane < 16) ? 0 : 8;
#pragma unroll
    for (int r=0;r<8;++r){
      const int m = mbase + r;
      float y = c[r] + bv;
      if (resid) y += resid[m*HH + n];
      out[m*HH + n] = y;
    }
  }
}

// ---- prep kernels ----
__global__ void tpad_kernel(const float* src, int Kin, int Nout, _Float16* dst, int Kpad){
  int idx = blockIdx.x*256 + threadIdx.x;
  int total = Nout*Kpad;
  if (idx >= total) return;
  int n = idx / Kpad, k = idx - n*Kpad;
  dst[idx] = (k < Kin) ? (_Float16)src[k*Nout + n] : (_Float16)0.0f;
}

__global__ void zero_kernel(float* p){ if (threadIdx.x < 8) p[threadIdx.x] = 0.f; }

__global__ void init_kernel(const float* pos, const float* vec, const float* scalars,
                            const int* roles, const float* t,
                            const float* nx, const float* nv, const float* ns,
                            const float* mrand,
                            float* x, float* v, float* s, int* rn){
  int i = blockIdx.x*256 + threadIdx.x;
  if (i >= NN) return;
  float tn = t[i / PP];
  float cv = cosf((tn + 0.008f) * (1.5707963267948966f / 1.008f));
  float a  = cv*cv;
  float sa = sqrtf(a);
  float sb = sqrtf(fmaxf(1.0f - a, 0.0f));
#pragma unroll
  for (int c=0;c<3;++c){
    x[i*3+c] = sa*pos[i*3+c] + sb*nx[i*3+c];
    v[i*3+c] = sa*vec[i*3+c] + sb*nv[i*3+c];
  }
#pragma unroll
  for (int q=0;q<NSS;++q) s[i*NSS+q] = sa*scalars[i*NSS+q] + sb*ns[i*NSS+q];
  int r = roles[i];
  rn[i] = ((mrand[i] < tn) || (r == NULLR)) ? NULLR : r;
}

__global__ void temb_kernel(const float* t, const float* w1, const float* b1,
                            const float* w2, const float* b2, float* te){
  __shared__ float hid[HH];
  int g = blockIdx.x, j = threadIdx.x;
  float tv = t[g];
  hid[j] = silu_f(tv * w1[j] + b1[j]);
  __syncthreads();
  float acc = b2[j];
  for (int q=0;q<HH;++q) acc += hid[q]*w2[q*HH + j];
  te[g*HH + j] = acc;
}

__global__ void knn_kernel(const float* x, int* col){
  __shared__ float xs[PP][3];
  int g = blockIdx.x, r = threadIdx.x;
  if (r < PP){
    xs[r][0] = x[(g*PP+r)*3+0];
    xs[r][1] = x[(g*PP+r)*3+1];
    xs[r][2] = x[(g*PP+r)*3+2];
  }
  __syncthreads();
  if (r < PP){
    float d2[PP];
    for (int j=0;j<PP;++j){
      float dx = xs[r][0]-xs[j][0], dy = xs[r][1]-xs[j][1], dz = xs[r][2]-xs[j][2];
      d2[j] = dx*dx + dy*dy + dz*dz;
    }
    d2[r] += 1.0e10f;
    for (int it=0; it<KK; ++it){
      float best = 3.0e38f; int bi = 0;
      for (int j=0;j<PP;++j) if (d2[j] < best){ best = d2[j]; bi = j; }
      col[(g*PP+r)*KK + it] = g*PP + bi;
      d2[bi] = 3.2e38f;
    }
  }
}

// ---- roots MLP: h = silu(h0 @ W1 + b1) @ W2 + b2, h0 = [s@s_w+s_b | role_emb | t_emb] (384) ----
#define RT_WAVES 2
__global__ void __launch_bounds__(RT_WAVES*32)
roots_kernel(const float* s, const int* rn, const float* te,
             const float* s_w, const float* s_b, const float* role_emb,
             const _Float16* w1T, const float* b1,
             const _Float16* w2T, const float* b2,
             float* hout){
  __shared__ __align__(16) _Float16 sA[RT_WAVES][16*384];
  __shared__ __align__(16) _Float16 sB[RT_WAVES][16*128];
  const int lane = threadIdx.x & 31, w = threadIdx.x >> 5;
  const int nb = (blockIdx.x*RT_WAVES + w) * 16;
  for (int idx = lane; idx < 16*384; idx += 32){
    int m = idx / 384, j = idx - m*384;
    int node = nb + m;
    float val;
    if (j < 128){
      float acc = s_b[j];
#pragma unroll
      for (int q=0;q<NSS;++q) acc += s[node*NSS+q]*s_w[q*HH+j];
      val = acc;
    } else if (j < 256){
      val = role_emb[rn[node]*HH + (j-128)];
    } else {
      val = te[(node/PP)*HH + (j-256)];
    }
    sA[w][idx] = (_Float16)val;
  }
  __syncthreads();
  gemm_lds(sA[w], 384, 12, w1T, 384, b1, true, sB[w], 128, lane);
  __syncthreads();
  gemm_glb(sB[w], 128, 4, w2T, 128, b2, nullptr, hout + nb*HH, lane);
}

// ---- message kernel: one wave = one node = 12 edges (rows 12-15 zero-padded) ----
#define MSG_WAVES 4
__global__ void __launch_bounds__(MSG_WAVES*32)
msg_kernel(const float* xcur, float* xnext, float* v,
           const float* hcur, float* aggr, const int* col,
           const _Float16* w1T, const float* b1,
           const _Float16* w2T, const float* b2,
           const _Float16* cw1T, const float* cb1, const float* cw2, const float* cb2,
           const _Float16* vw1T, const float* vb1, const float* vw2, const float* vb2){
  __shared__ __align__(16) _Float16 sA[MSG_WAVES][16*288];
  __shared__ __align__(16) _Float16 sB[MSG_WAVES][16*128];
  __shared__ float sRel[MSG_WAVES][16][4];
  __shared__ float sW[MSG_WAVES][32];
  __shared__ int   sCol[MSG_WAVES][16];

  const int lane = threadIdx.x & 31, w = threadIdx.x >> 5;
  const int node = blockIdx.x*MSG_WAVES + w;

  const float xi0 = xcur[node*3+0], xi1 = xcur[node*3+1], xi2 = xcur[node*3+2];

  if (lane < 16){
    int k = lane;
    if (k < KK){
      int c = col[node*KK + k];
      sCol[w][k] = c;
      float r0 = xi0 - xcur[c*3+0];
      float r1 = xi1 - xcur[c*3+1];
      float r2 = xi2 - xcur[c*3+2];
      sRel[w][k][0]=r0; sRel[w][k][1]=r1; sRel[w][k][2]=r2;
      sRel[w][k][3]=r0*r0+r1*r1+r2*r2;
    } else {
      sCol[w][k] = node;
      sRel[w][k][0]=0.f; sRel[w][k][1]=0.f; sRel[w][k][2]=0.f; sRel[w][k][3]=0.f;
    }
  }
  __syncthreads();

  // stage m_in tile (16 x 288): [h_src(128) | h_nbr(128) | sq | zd | zero-pad]
  for (int idx = lane; idx < 16*288; idx += 32){
    int k = idx / 288, j = idx - k*288;
    float val;
    if (k >= KK)        val = 0.f;
    else if (j < 128)   val = hcur[node*HH + j];
    else if (j < 256)   val = hcur[sCol[w][k]*HH + (j-128)];
    else if (j == 256)  val = sRel[w][k][3];
    else if (j == 257)  val = sRel[w][k][2];
    else                val = 0.f;
    sA[w][idx] = (_Float16)val;
  }
  __syncthreads();

  gemm_lds(sA[w], 288, 9, w1T, 288, b1, true, sB[w], 128, lane);   // hid
  __syncthreads();
  gemm_lds(sB[w], 128, 4, w2T, 128, b2, true, sA[w], 288, lane);   // msg (cols 0..127 of sA)
  __syncthreads();

  // aggr = sum over the node's 12 edges
  for (int n = lane; n < HH; n += 32){
    float acc = 0.f;
    for (int k=0;k<KK;++k) acc += (float)sA[w][k*288 + n];
    aggr[node*HH + n] = acc;
  }
  gemm_lds(sA[w], 288, 4, cw1T, 128, cb1, true, sB[w], 128, lane);       // c-head hidden
  gemm_lds(sA[w], 288, 4, vw1T, 128, vb1, true, sA[w]+160, 288, lane);   // v-head hidden
  __syncthreads();

  { // head dots (128 -> 1), lanes 0-15: c-head, lanes 16-31: v-head
    int k = lane & 15;
    if (k < KK){
      float d = 0.f;
      if (lane < 16){
        for (int j=0;j<HH;++j) d += (float)sB[w][k*128+j] * cw2[j];
        d += cb2[0];
        sW[w][k] = d / (sRel[w][k][3] + 1e-8f);
      } else {
        for (int j=0;j<HH;++j) d += (float)sA[w][k*288+160+j] * vw2[j];
        d += vb2[0];
        sW[w][16+k] = d / (sRel[w][k][3] + 1e-8f);
      }
    } else {
      if (lane < 16) sW[w][k] = 0.f; else sW[w][16+k] = 0.f;
    }
  }
  __syncthreads();

  if (lane < 3){
    float dx = 0.f, dv = 0.f;
    for (int k=0;k<KK;++k){
      dx += sRel[w][k][lane] * sW[w][k];
      dv += sRel[w][k][lane] * sW[w][16+k];
    }
    xnext[node*3+lane] = xcur[node*3+lane] + dx;
    v[node*3+lane]    += dv;
  }
}

// ---- node update: h_next = h + silu([h|aggr] @ W1 + b1) @ W2 + b2 ----
#define ND_WAVES 4
__global__ void __launch_bounds__(ND_WAVES*32)
node_kernel(const float* hcur, float* hnext, const float* aggr,
            const _Float16* w1T, const float* b1,
            const _Float16* w2T, const float* b2){
  __shared__ __align__(16) _Float16 sA[ND_WAVES][16*256];
  __shared__ __align__(16) _Float16 sB[ND_WAVES][16*128];
  const int lane = threadIdx.x & 31, w = threadIdx.x >> 5;
  const int nb = (blockIdx.x*ND_WAVES + w) * 16;
  for (int idx = lane; idx < 16*256; idx += 32){
    int m = idx >> 8, j = idx & 255;
    float val = (j < 128) ? hcur[(nb+m)*HH + j] : aggr[(nb+m)*HH + (j-128)];
    sA[w][idx] = (_Float16)val;
  }
  __syncthreads();
  gemm_lds(sA[w], 256, 8, w1T, 256, b1, true, sB[w], 128, lane);
  __syncthreads();
  gemm_glb(sB[w], 128, 4, w2T, 128, b2, hcur + nb*HH, hnext + nb*HH, lane);
}

// ---- losses ----
__global__ void loss_kernel(const float* x, const float* v, const float* h,
                            const float* pos, const float* vec, const float* scalars,
                            const int* roles,
                            const float* fh_w, const float* fh_b,
                            const float* rh_w, const float* rh_b,
                            float* acc){
  __shared__ float red[256*4];
  const int tid = threadIdx.x;
  const int i = blockIdx.x*256 + tid;
  const int role = roles[i];
  const float isr = (role != NULLR) ? 1.f : 0.f;
  float xs=0.f, vs=0.f, ss=0.f, rs=0.f;
#pragma unroll
  for (int c=0;c<3;++c){
    float dx = x[i*3+c] - pos[i*3+c]; xs += isr*dx*dx;
    float dv = v[i*3+c] - vec[i*3+c]; vs += isr*dv*dv;
  }
  for (int q=0;q<NSS;++q){
    float val = fh_b[q];
    for (int j=0;j<HH;++j) val += h[i*HH+j]*fh_w[j*NSS+q];
    float d = val - scalars[i*NSS+q];
    ss += isr*d*d;
  }
  float lg[NRR];
  for (int r=0;r<NRR;++r){
    float val = rh_b[r];
    for (int j=0;j<HH;++j) val += h[i*HH+j]*rh_w[j*NRR+r];
    lg[r] = val;
  }
  float m = lg[0];
  for (int r=1;r<NRR;++r) m = fmaxf(m, lg[r]);
  float sum = 0.f;
  for (int r=0;r<NRR;++r) sum += __expf(lg[r]-m);
  float lse = m + __logf(sum);
  rs = -(lg[role] - lse);

  red[tid*4+0]=xs; red[tid*4+1]=vs; red[tid*4+2]=ss; red[tid*4+3]=rs;
  __syncthreads();
  for (int st=128; st>0; st>>=1){
    if (tid < st){
#pragma unroll
      for (int c=0;c<4;++c) red[tid*4+c] += red[(tid+st)*4+c];
    }
    __syncthreads();
  }
  if (tid == 0){
#pragma unroll
    for (int c=0;c<4;++c) atomicAdd(&acc[c], red[c]);
  }
}

__global__ void fin_kernel(const float* acc, float* out){
  float xl = acc[0] / (float)(NN*3);
  float vl = acc[1] / (float)(NN*3);
  float sl = acc[2] / (float)(NN*NSS);
  float rl = acc[3] / (float)NN;
  out[0]=xl; out[1]=vl; out[2]=sl; out[3]=rl; out[4]=xl+vl+sl+rl;
}

// ---- host ----
extern "C" void kernel_launch(void* const* d_in, const int* in_sizes, int n_in,
                              void* d_out, int out_size, void* d_ws, size_t ws_size,
                              hipStream_t stream){
  (void)n_in; (void)out_size; (void)ws_size;
  // params: insertion order vs jax-sorted pytree order, auto-detect
  static const int sortedPos[31] = {25,23,26,24,16,22,21,19,17,20,18,8,6,9,7,12,10,13,11,2,0,3,1,29,27,30,28,5,4,15,14};
  const bool sorted = (in_sizes[0] != 128);
  const float* prm[31];
  for (int i=0;i<31;++i) prm[i] = (const float*)d_in[ sorted ? sortedPos[i] : i ];
  // insertion-order ids:
  // 0 t_w1, 1 t_b1, 2 t_w2, 3 t_b2, 4 role_emb, 5 s_w, 6 s_b, 7 roots_w1, 8 roots_b1,
  // 9 roots_w2, 10 roots_b2, 11 msg_w1, 12 msg_b1, 13 msg_w2, 14 msg_b2, 15 node_w1,
  // 16 node_b1, 17 node_w2, 18 node_b2, 19 c_w1, 20 c_b1, 21 c_w2, 22 c_b2,
  // 23 v_w1, 24 v_b1, 25 v_w2, 26 v_b2, 27 fh_w, 28 fh_b, 29 rh_w, 30 rh_b
  const float* pos      = (const float*)d_in[31];
  const float* vec      = (const float*)d_in[32];
  const float* scalars  = (const float*)d_in[33];
  const int*   roles    = (const int*)  d_in[34];
  const float* tptr     = (const float*)d_in[35];
  const float* noise_x  = (const float*)d_in[36];
  const float* noise_v  = (const float*)d_in[37];
  const float* noise_s  = (const float*)d_in[38];
  const float* mrand    = (const float*)d_in[39];

  char* wsp = (char*)d_ws;
  size_t off = 0;
  auto carve = [&](size_t bytes){ void* p = wsp + off; off += (bytes + 255) & ~(size_t)255; return p; };
  float* xb0  = (float*)carve((size_t)NN*3*4);
  float* xb1  = (float*)carve((size_t)NN*3*4);
  float* vb   = (float*)carve((size_t)NN*3*4);
  float* sb   = (float*)carve((size_t)NN*NSS*4);
  float* hb0  = (float*)carve((size_t)NN*HH*4);
  float* hb1  = (float*)carve((size_t)NN*HH*4);
  float* ag   = (float*)carve((size_t)NN*HH*4);
  float* te   = (float*)carve((size_t)GG*HH*4);
  int*   col  = (int*)  carve((size_t)NN*KK*4);
  int*   rn   = (int*)  carve((size_t)NN*4);
  float* lacc = (float*)carve(8*4);
  _Float16* wp = (_Float16*)carve((size_t)876544*2);

  auto tp = [&](const float* src, int Kin, int Nout, _Float16* dst, int Kpad){
    int total = Nout*Kpad;
    tpad_kernel<<<(total+255)/256, 256, 0, stream>>>(src, Kin, Nout, dst, Kpad);
  };
  _Float16* rootsW1T = wp;
  _Float16* rootsW2T = wp + 49152;
  _Float16* lbase    = wp + 65536;
  tp(prm[7], 384, 128, rootsW1T, 384);
  tp(prm[9], 128, 128, rootsW2T, 128);
  for (int l=0;l<LL;++l){
    _Float16* b = lbase + (size_t)l*135168;
    tp(prm[11] + (size_t)l*258*128, 258, 128, b + 0,      288);  // msg_w1T
    tp(prm[13] + (size_t)l*128*128, 128, 128, b + 36864,  128);  // msg_w2T
    tp(prm[19] + (size_t)l*128*128, 128, 128, b + 53248,  128);  // c_w1T
    tp(prm[23] + (size_t)l*128*128, 128, 128, b + 69632,  128);  // v_w1T
    tp(prm[15] + (size_t)l*256*128, 256, 128, b + 86016,  256);  // node_w1T
    tp(prm[17] + (size_t)l*128*128, 128, 128, b + 118784, 128);  // node_w2T
  }

  zero_kernel<<<1, 32, 0, stream>>>(lacc);
  init_kernel<<<NN/256, 256, 0, stream>>>(pos, vec, scalars, roles, tptr,
                                          noise_x, noise_v, noise_s, mrand,
                                          xb0, vb, sb, rn);
  temb_kernel<<<GG, HH, 0, stream>>>(tptr, prm[0], prm[1], prm[2], prm[3], te);
  knn_kernel<<<GG, 32, 0, stream>>>(xb0, col);
  roots_kernel<<<(NN/16)/RT_WAVES, RT_WAVES*32, 0, stream>>>(sb, rn, te,
      prm[5], prm[6], prm[4], rootsW1T, prm[8], rootsW2T, prm[10], hb0);

  for (int l=0;l<LL;++l){
    _Float16* b = lbase + (size_t)l*135168;
    const float* xc = (l & 1) ? xb1 : xb0;
    float*       xn = (l & 1) ? xb0 : xb1;
    const float* hc = (l & 1) ? hb1 : hb0;
    float*       hn = (l & 1) ? hb0 : hb1;
    msg_kernel<<<NN/MSG_WAVES, MSG_WAVES*32, 0, stream>>>(xc, xn, vb, hc, ag, col,
        b + 0,      prm[12] + l*HH,
        b + 36864,  prm[14] + l*HH,
        b + 53248,  prm[20] + l*HH, prm[21] + l*HH, prm[22] + l,
        b + 69632,  prm[24] + l*HH, prm[25] + l*HH, prm[26] + l);
    node_kernel<<<(NN/16)/ND_WAVES, ND_WAVES*32, 0, stream>>>(hc, hn, ag,
        b + 86016,  prm[16] + l*HH,
        b + 118784, prm[18] + l*HH);
  }

  loss_kernel<<<NN/256, 256, 0, stream>>>(xb0, vb, hb0, pos, vec, scalars, roles,
                                          prm[27], prm[28], prm[29], prm[30], lacc);
  fin_kernel<<<1, 1, 0, stream>>>(lacc, (float*)d_out);
}
